// AtomRepresentationModel_55568286875775
// MI455X (gfx1250) — matrix operand order repaired and checked by reading, not compiled, not run on hardware
//
#include <hip/hip_runtime.h>
#include <hip/hip_bf16.h>

// ---------------- problem constants (from reference) ----------------
constexpr int kB      = 8;
constexpr int kNper   = 2048;
constexpr int kEper   = 32768;
constexpr int kH      = 128;            // hidden size
constexpr int kL      = 3;              // interaction layers
constexpr int kNtot   = kB * kNper;     // 16384
constexpr int kEtot   = kB * kEper;     // 262144
constexpr int kES     = 50;             // gaussian basis size
constexpr int kESpad  = 64;             // padded K for WMMA
constexpr float kLOG2 = 0.69314718055994530942f;

// WMMA tiling
constexpr int kNW  = 4;                 // waves per block (wave32)
constexpr int kLDH = 272;               // LDS row stride in halves (256 + pad)

typedef __attribute__((ext_vector_type(16))) _Float16 v16h;
typedef __attribute__((ext_vector_type(8)))  float    v8f;

// shifted softplus
__device__ __forceinline__ float sspf(float x) {
  float sp = (x > 15.0f) ? x : logf(1.0f + expf(x));
  return sp - kLOG2;
}

// A fragment (16x32 f16) from an LDS row-major stage, per ISA 7.12.2:
// lanes 0-15: M=lane, K = {0..7, 16..23}+kbase ; lanes 16-31: M=lane-16, K = {8..15, 24..31}+kbase
__device__ __forceinline__ v16h load_a_frag(const _Float16* S, int lane, int kbase) {
  const int m = lane & 15;
  const _Float16* p = S + m * kLDH + kbase + ((lane >> 4) << 3);
  v16h a;
#pragma unroll
  for (int j = 0; j < 8; ++j) a[j] = p[j];
#pragma unroll
  for (int j = 0; j < 8; ++j) a[j + 8] = p[j + 16];
  return a;
}

// B fragment (32x16 f16) from pre-packed global weights: 16 contiguous halves per lane.
__device__ __forceinline__ v16h load_b_frag(const _Float16* pack, int lane, int kt, int nt) {
  return *(const v16h*)(pack + ((((kt << 3) + nt) << 5 | lane) << 4));
}

__device__ __forceinline__ v8f wmma_step(v16h a, v16h b, v8f c) {
  return __builtin_amdgcn_wmma_f32_16x16x32_f16(false, a, false, b, (short)0, c, false, false);
}

// ---------------- one-time prep kernels ----------------

// Pack fp32 weight W[K x 128] into WMMA-B fragment order (f16), zero-padding K to ktiles*32.
__global__ void pack_kernel(const float* __restrict__ W, _Float16* __restrict__ out,
                            int Kreal, int ktiles) {
  int idx = blockIdx.x * blockDim.x + threadIdx.x;
  int total = ktiles * 4096;            // ktiles * 8 ntiles * 32 lanes * 16 halves
  if (idx >= total) return;
  int j  = idx & 15;
  int l  = (idx >> 4) & 31;
  int nt = (idx >> 9) & 7;
  int kt = idx >> 12;
  int k  = kt * 32 + ((l >= 16) ? 8 : 0) + ((j >= 8) ? 8 : 0) + j;
  int n  = nt * 16 + (l & 15);
  float v = (k < Kreal) ? W[k * kH + n] : 0.0f;
  out[idx] = (_Float16)v;
}

__global__ void init_nodes_kernel(const int* __restrict__ nodes, const float* __restrict__ emb,
                                  float* __restrict__ node_state) {
  int idx = blockIdx.x * blockDim.x + threadIdx.x;   // N_tot * H threads
  int n = idx >> 7, c = idx & 127;
  node_state[idx] = emb[nodes[n] * kH + c];
}

__global__ void edge_flatten_kernel(const int* __restrict__ ae,
                                    int* __restrict__ snd, int* __restrict__ rcv) {
  int e = blockIdx.x * blockDim.x + threadIdx.x;
  if (e >= kEtot) return;
  int b = e / kEper;                                 // per-graph node offset = b*2048
  snd[e] = ae[2 * e + 0] + b * kNper;
  rcv[e] = ae[2 * e + 1] + b * kNper;
}

// ---------------- fused edge-message kernel (one layer) ----------------
// 4 waves/block, 16 edges per wave. Pipeline:
//   gather pair -> WMMA K=256 -> ssp -> WMMA K=128  (m_node, kept in regs)
//   gaussian basis -> WMMA K=64 -> ssp -> WMMA K=128 (gates)
//   msg = m_node * gates * cutoff ; atomicAdd into msum[receiver]
__global__ __launch_bounds__(128) void edge_kernel(
    const int* __restrict__ snd, const int* __restrict__ rcv,
    const float* __restrict__ dist, const float* __restrict__ node_state,
    const _Float16* __restrict__ Wn1p, const float* __restrict__ bn1,
    const _Float16* __restrict__ Wn2p, const float* __restrict__ bn2,
    const _Float16* __restrict__ We1p, const float* __restrict__ be1,
    const _Float16* __restrict__ We2p, const float* __restrict__ be2,
    float* __restrict__ msum) {
  __shared__ _Float16 stA[kNW][16 * kLDH];
  const int lane  = threadIdx.x & 31;
  const int w     = threadIdx.x >> 5;
  const int e0    = (blockIdx.x * kNW + w) * 16;
  _Float16* S     = stA[w];
  const int mcol  = lane & 15;
  const int rbase = (lane >> 4) << 3;

  // ---- stage A = [sender_feat || receiver_feat], fp32 -> f16 (2 lanes per edge row)
  {
    int row = lane >> 1, half = lane & 1;
    int e = e0 + row;
    int node = half ? rcv[e] : snd[e];
    const float4* src = (const float4*)(node_state + (size_t)node * kH);
    _Float16* dstp = S + row * kLDH + half * kH;
#pragma unroll
    for (int c = 0; c < kH / 4; ++c) {
      float4 v = src[c];
      dstp[4 * c + 0] = (_Float16)v.x; dstp[4 * c + 1] = (_Float16)v.y;
      dstp[4 * c + 2] = (_Float16)v.z; dstp[4 * c + 3] = (_Float16)v.w;
    }
  }

  v8f accM[8];
  // GEMM1: [16x256] @ Wn1 + bn1, then ssp -> LDS
#pragma unroll
  for (int nt = 0; nt < 8; ++nt) {
    float bv = bn1[nt * 16 + mcol];
#pragma unroll
    for (int j = 0; j < 8; ++j) accM[nt][j] = bv;
  }
#pragma unroll
  for (int kt = 0; kt < 8; ++kt) {
    v16h a = load_a_frag(S, lane, kt * 32);
#pragma unroll
    for (int nt = 0; nt < 8; ++nt)
      accM[nt] = wmma_step(a, load_b_frag(Wn1p, lane, kt, nt), accM[nt]);
  }
#pragma unroll
  for (int nt = 0; nt < 8; ++nt)
#pragma unroll
    for (int r = 0; r < 8; ++r)
      S[(r + rbase) * kLDH + nt * 16 + mcol] = (_Float16)sspf(accM[nt][r]);

  // GEMM2: [16x128] @ Wn2 + bn2 -> m_node (kept in accM)
#pragma unroll
  for (int nt = 0; nt < 8; ++nt) {
    float bv = bn2[nt * 16 + mcol];
#pragma unroll
    for (int j = 0; j < 8; ++j) accM[nt][j] = bv;
  }
#pragma unroll
  for (int kt = 0; kt < 4; ++kt) {
    v16h a = load_a_frag(S, lane, kt * 32);
#pragma unroll
    for (int nt = 0; nt < 8; ++nt)
      accM[nt] = wmma_step(a, load_b_frag(Wn2p, lane, kt, nt), accM[nt]);
  }

  // ---- gates path: gaussian radial basis (sigma=0.1), K padded to 64
  {
    int row = lane >> 1, half = lane & 1;
    float d = dist[e0 + row];
    _Float16* dstp = S + row * kLDH + half * 32;
#pragma unroll
    for (int c = 0; c < 32; ++c) {
      int col = half * 32 + c;
      float mu = 0.1f * (float)col;
      float t  = d - mu;
      float v  = (col < kES) ? expf(-50.0f * t * t) : 0.0f;
      dstp[c] = (_Float16)v;
    }
  }

  v8f accG[8];
#pragma unroll
  for (int nt = 0; nt < 8; ++nt) {
    float bv = be1[nt * 16 + mcol];
#pragma unroll
    for (int j = 0; j < 8; ++j) accG[nt][j] = bv;
  }
#pragma unroll
  for (int kt = 0; kt < 2; ++kt) {
    v16h a = load_a_frag(S, lane, kt * 32);
#pragma unroll
    for (int nt = 0; nt < 8; ++nt)
      accG[nt] = wmma_step(a, load_b_frag(We1p, lane, kt, nt), accG[nt]);
  }
#pragma unroll
  for (int nt = 0; nt < 8; ++nt)
#pragma unroll
    for (int r = 0; r < 8; ++r)
      S[(r + rbase) * kLDH + nt * 16 + mcol] = (_Float16)sspf(accG[nt][r]);

#pragma unroll
  for (int nt = 0; nt < 8; ++nt) {
    float bv = be2[nt * 16 + mcol];
#pragma unroll
    for (int j = 0; j < 8; ++j) accG[nt][j] = bv;
  }
#pragma unroll
  for (int kt = 0; kt < 4; ++kt) {
    v16h a = load_a_frag(S, lane, kt * 32);
#pragma unroll
    for (int nt = 0; nt < 8; ++nt)
      accG[nt] = wmma_step(a, load_b_frag(We2p, lane, kt, nt), accG[nt]);
  }

  // ---- messages = m_node * gates * cutoff ; segment-sum via f32 atomics
#pragma unroll
  for (int r = 0; r < 8; ++r) {
    int e = e0 + r + rbase;
    float d = dist[e];
    float cut = 1.0f / (1.0f + expf(5.0f * (d - 3.5f)));   // 1 - sigmoid(5*(d-3.5))
    float* dst = msum + (size_t)rcv[e] * kH + mcol;
#pragma unroll
    for (int nt = 0; nt < 8; ++nt)
      atomicAdd(dst + nt * 16, accM[nt][r] * accG[nt][r] * cut);
  }
}

// ---------------- node state-transition kernel (one layer) ----------------
__global__ __launch_bounds__(128) void node_update_kernel(
    const float* __restrict__ msum,
    const _Float16* __restrict__ Ws1p, const float* __restrict__ bs1,
    const _Float16* __restrict__ Ws2p, const float* __restrict__ bs2,
    float* __restrict__ node_state, float* __restrict__ out) {
  __shared__ _Float16 stA[kNW][16 * kLDH];
  const int lane  = threadIdx.x & 31;
  const int w     = threadIdx.x >> 5;
  const int n0    = (blockIdx.x * kNW + w) * 16;
  _Float16* S     = stA[w];
  const int mcol  = lane & 15;
  const int rbase = (lane >> 4) << 3;

  {
    int row = lane >> 1, half = lane & 1;
    const float4* src = (const float4*)(msum + (size_t)(n0 + row) * kH + half * 64);
    _Float16* dstp = S + row * kLDH + half * 64;
#pragma unroll
    for (int c = 0; c < 16; ++c) {
      float4 v = src[c];
      dstp[4 * c + 0] = (_Float16)v.x; dstp[4 * c + 1] = (_Float16)v.y;
      dstp[4 * c + 2] = (_Float16)v.z; dstp[4 * c + 3] = (_Float16)v.w;
    }
  }

  v8f acc[8];
#pragma unroll
  for (int nt = 0; nt < 8; ++nt) {
    float bv = bs1[nt * 16 + mcol];
#pragma unroll
    for (int j = 0; j < 8; ++j) acc[nt][j] = bv;
  }
#pragma unroll
  for (int kt = 0; kt < 4; ++kt) {
    v16h a = load_a_frag(S, lane, kt * 32);
#pragma unroll
    for (int nt = 0; nt < 8; ++nt)
      acc[nt] = wmma_step(a, load_b_frag(Ws1p, lane, kt, nt), acc[nt]);
  }
#pragma unroll
  for (int nt = 0; nt < 8; ++nt)
#pragma unroll
    for (int r = 0; r < 8; ++r)
      S[(r + rbase) * kLDH + nt * 16 + mcol] = (_Float16)sspf(acc[nt][r]);

#pragma unroll
  for (int nt = 0; nt < 8; ++nt) {
    float bv = bs2[nt * 16 + mcol];
#pragma unroll
    for (int j = 0; j < 8; ++j) acc[nt][j] = bv;
  }
#pragma unroll
  for (int kt = 0; kt < 4; ++kt) {
    v16h a = load_a_frag(S, lane, kt * 32);
#pragma unroll
    for (int nt = 0; nt < 8; ++nt)
      acc[nt] = wmma_step(a, load_b_frag(Ws2p, lane, kt, nt), acc[nt]);
  }

#pragma unroll
  for (int nt = 0; nt < 8; ++nt)
#pragma unroll
    for (int r = 0; r < 8; ++r) {
      size_t ix = (size_t)(n0 + r + rbase) * kH + nt * 16 + mcol;
      float v = acc[nt][r] + node_state[ix];   // residual
      node_state[ix] = v;
      out[ix] = v;
    }
}

// ---------------- launcher ----------------
extern "C" void kernel_launch(void* const* d_in, const int* in_sizes, int n_in,
                              void* d_out, int out_size, void* d_ws, size_t ws_size,
                              hipStream_t stream) {
  (void)in_sizes; (void)n_in; (void)out_size; (void)ws_size;
  const int*   nodes = (const int*)d_in[0];
  const int*   ae    = (const int*)d_in[1];
  const float* dist  = (const float*)d_in[2];
  const float* emb   = (const float*)d_in[5];
  const float* Wn1 = (const float*)d_in[6];  const float* bn1 = (const float*)d_in[7];
  const float* Wn2 = (const float*)d_in[8];  const float* bn2 = (const float*)d_in[9];
  const float* We1 = (const float*)d_in[10]; const float* be1 = (const float*)d_in[11];
  const float* We2 = (const float*)d_in[12]; const float* be2 = (const float*)d_in[13];
  const float* Ws1 = (const float*)d_in[14]; const float* bs1 = (const float*)d_in[15];
  const float* Ws2 = (const float*)d_in[16]; const float* bs2 = (const float*)d_in[17];
  float* out = (float*)d_out;

  // workspace carve-up
  char* ws = (char*)d_ws; size_t off = 0;
  auto carve = [&](size_t bytes) { void* p = ws + off; off = (off + bytes + 255) & ~(size_t)255; return p; };
  float*    node_state = (float*)   carve((size_t)kNtot * kH * 4);
  float*    msum       = (float*)   carve((size_t)kNtot * kH * 4);
  int*      snd        = (int*)     carve((size_t)kEtot * 4);
  int*      rcv        = (int*)     carve((size_t)kEtot * 4);
  _Float16* Wn1p = (_Float16*)carve((size_t)kL * 8 * 4096 * 2);  // K=256: 8 ktiles
  _Float16* Wn2p = (_Float16*)carve((size_t)kL * 4 * 4096 * 2);  // K=128: 4 ktiles
  _Float16* We1p = (_Float16*)carve((size_t)kL * 2 * 4096 * 2);  // K=64 : 2 ktiles
  _Float16* We2p = (_Float16*)carve((size_t)kL * 4 * 4096 * 2);
  _Float16* Ws1p = (_Float16*)carve((size_t)kL * 4 * 4096 * 2);
  _Float16* Ws2p = (_Float16*)carve((size_t)kL * 4 * 4096 * 2);

  // one-time prep (re-done each call; deterministic)
  for (int l = 0; l < kL; ++l) {
    pack_kernel<<<(8 * 4096 + 255) / 256, 256, 0, stream>>>(Wn1 + (size_t)l * 256 * kH, Wn1p + (size_t)l * 8 * 4096, 256, 8);
    pack_kernel<<<(4 * 4096 + 255) / 256, 256, 0, stream>>>(Wn2 + (size_t)l * 128 * kH, Wn2p + (size_t)l * 4 * 4096, 128, 4);
    pack_kernel<<<(2 * 4096 + 255) / 256, 256, 0, stream>>>(We1 + (size_t)l * kES * kH, We1p + (size_t)l * 2 * 4096, kES, 2);
    pack_kernel<<<(4 * 4096 + 255) / 256, 256, 0, stream>>>(We2 + (size_t)l * 128 * kH, We2p + (size_t)l * 4 * 4096, 128, 4);
    pack_kernel<<<(4 * 4096 + 255) / 256, 256, 0, stream>>>(Ws1 + (size_t)l * 128 * kH, Ws1p + (size_t)l * 4 * 4096, 128, 4);
    pack_kernel<<<(4 * 4096 + 255) / 256, 256, 0, stream>>>(Ws2 + (size_t)l * 128 * kH, Ws2p + (size_t)l * 4 * 4096, 128, 4);
  }
  init_nodes_kernel<<<(kNtot * kH) / 256, 256, 0, stream>>>(nodes, emb, node_state);
  edge_flatten_kernel<<<kEtot / 256, 256, 0, stream>>>(ae, snd, rcv);

  for (int l = 0; l < kL; ++l) {
    hipMemsetAsync(msum, 0, (size_t)kNtot * kH * 4, stream);
    edge_kernel<<<kEtot / (kNW * 16), kNW * 32, 0, stream>>>(
        snd, rcv, dist, node_state,
        Wn1p + (size_t)l * 8 * 4096, bn1 + l * kH,
        Wn2p + (size_t)l * 4 * 4096, bn2 + l * kH,
        We1p + (size_t)l * 2 * 4096, be1 + l * kH,
        We2p + (size_t)l * 4 * 4096, be2 + l * kH, msum);
    node_update_kernel<<<kNtot / (kNW * 16), kNW * 32, 0, stream>>>(
        msum,
        Ws1p + (size_t)l * 4 * 4096, bs1 + l * kH,
        Ws2p + (size_t)l * 4 * 4096, bs2 + l * kH,
        node_state, out + (size_t)l * kNtot * kH);
  }
}
